// SparseNeuralDecisionTreeHead_12137577578953
// MI455X (gfx1250) — compile-verified, matrix operand order checked by reading, not run on hardware
//
#include <hip/hip_runtime.h>

#define DEPTH        4
#define NUM_INTERNAL 15
#define NUM_LEAVES   16
#define INPUT_DIM    1024
#define NUM_CLASSES  100
#define LEAF_HIDDEN  256
#define BATCH        4096
#define NPAD         128      // NUM_CLASSES padded to 8 x 16 column tiles
#define EPSG         1e-10f

// LDS A-tile staging: 64 rows x 64 k (bf16), row padded 64->72 elements
#define SROW         72
#define CHUNK_SHORTS (64 * SROW)          // per buffer, in shorts
#define CHUNK_BYTES  (CHUNK_SHORTS * 2)   // 9216

typedef __attribute__((ext_vector_type(16))) __bf16 bf16x16;
typedef __attribute__((ext_vector_type(8)))  float  f32x8;

union Frag16 { bf16x16 v; uint4 q[2]; };

__device__ __forceinline__ unsigned short f2bf(float f) {
    unsigned int u = __float_as_uint(f);
    u += 0x7FFFu + ((u >> 16) & 1u);      // round-to-nearest-even
    return (unsigned short)(u >> 16);
}

// gfx1250 async copy: each lane moves 16B global -> LDS, tracked by ASYNCcnt.
__device__ __forceinline__ void async_copy_b128(unsigned lds_byte_addr,
                                                const unsigned short* gaddr) {
    asm volatile("global_load_async_to_lds_b128 %0, %1, off"
                 :: "v"(lds_byte_addr), "v"(gaddr)
                 : "memory");
}

// Stage one 64x64 bf16 chunk (rows start at gbase, row stride LDA elements)
// into LDS at ldsbase. 256 threads x 2 units of 16B each = 8 KB payload,
// 2 async instructions per wave.
template <int LDA>
__device__ __forceinline__ void stage_chunk(const unsigned short* __restrict__ gbase,
                                            unsigned ldsbase, int tid) {
    #pragma unroll
    for (int j = 0; j < 2; ++j) {
        const int i   = tid + 256 * j;        // 0..511
        const int row = i >> 3;               // 0..63
        const int u   = i & 7;                // 8 x 16B per row
        async_copy_b128(ldsbase + row * (SROW * 2) + u * 16,
                        gbase + (size_t)row * LDA + u * 8);
    }
}

// ---------------------------------------------------------------------------
// Kernel 0: precompute efl = exp(feature_logits), fwefl = fw * efl  [15][1024]
// ---------------------------------------------------------------------------
__global__ void sndt_prep_kernel(const float* __restrict__ flog,
                                 const float* __restrict__ fw,
                                 float* __restrict__ efl,
                                 float* __restrict__ fwefl) {
    const int i = blockIdx.x * blockDim.x + threadIdx.x;
    if (i >= NUM_INTERNAL * INPUT_DIM) return;
    const float e = __expf(flog[i]);
    efl[i]   = e;
    fwefl[i] = fw[i] * e;
}

// ---------------------------------------------------------------------------
// Kernel 1: Gumbel-softmax gating. One wave32 per (node, batch) row of 1024.
//   exp(fl + g) = exp(fl) / t  with t = eps - log(u + eps)
//   agg = sum(x*fw*efl/t) / sum(efl/t);  split_prob = sigmoid(agg - thr)
// Hot loop: 1 v_log + 1 v_rcp per element (vs 3 trans naive).
// ---------------------------------------------------------------------------
__global__ __launch_bounds__(256)
void sndt_gating_kernel(const float* __restrict__ x,
                        const float* __restrict__ gu,
                        const float* __restrict__ efl,
                        const float* __restrict__ thr,
                        const float* __restrict__ fwefl,
                        float* __restrict__ split_prob) {
    const int lane = threadIdx.x & 31;
    const int wave = threadIdx.x >> 5;
    const int gw   = blockIdx.x * 8 + wave;       // global row id
    const int n    = gw >> 12;                    // gw / 4096
    const int b    = gw & 4095;

    const float* gurow = gu + ((size_t)n * BATCH + b) * INPUT_DIM;
    const float* xrow  = x  + (size_t)b * INPUT_DIM;
    const float* erow  = efl   + n * INPUT_DIM;
    const float* wrow  = fwefl + n * INPUT_DIM;

    float s1 = 0.f, s2 = 0.f;
#define GCOMP(uu, xx, ee, ww)                                              \
    { float t   = EPSG - __logf((uu) + EPSG);                              \
      float inv = __builtin_amdgcn_rcpf(t);                                \
      s1 += (ee) * inv; s2 += (xx) * (ww) * inv; }

    for (int j = 0; j < 8; ++j) {
        const int d = j * 128 + lane * 4;
        __builtin_prefetch(gurow + d + 1024, 0, 1);  // global_prefetch_b8
        float4 u4 = *(const float4*)(gurow + d);
        float4 x4 = *(const float4*)(xrow  + d);
        float4 e4 = *(const float4*)(erow  + d);
        float4 w4 = *(const float4*)(wrow  + d);
        GCOMP(u4.x, x4.x, e4.x, w4.x)
        GCOMP(u4.y, x4.y, e4.y, w4.y)
        GCOMP(u4.z, x4.z, e4.z, w4.z)
        GCOMP(u4.w, x4.w, e4.w, w4.w)
    }
#undef GCOMP

    #pragma unroll
    for (int m = 16; m > 0; m >>= 1) {
        s1 += __shfl_xor(s1, m, 32);
        s2 += __shfl_xor(s2, m, 32);
    }
    if (lane == 0) {
        float agg = s2 / s1;
        float z   = agg - thr[n];
        split_prob[n * BATCH + b] = 1.f / (1.f + __expf(-z));
    }
}

// ---------------------------------------------------------------------------
// Kernel 2: leaf routing probabilities (heap-order traversal).
// ---------------------------------------------------------------------------
__global__ void sndt_leafprob_kernel(const float* __restrict__ sp,
                                     float* __restrict__ lp) {
    const int b = blockIdx.x * blockDim.x + threadIdx.x;
    if (b >= BATCH) return;
    float s[NUM_INTERNAL];
    #pragma unroll
    for (int i = 0; i < NUM_INTERNAL; ++i) s[i] = sp[i * BATCH + b];
    #pragma unroll
    for (int l = 0; l < NUM_LEAVES; ++l) {
        float p = 1.f;
        #pragma unroll
        for (int lev = 0; lev < DEPTH; ++lev) {
            const int prefix = l >> (DEPTH - lev);
            const int node   = (1 << lev) - 1 + prefix;
            const int bit    = (l >> (DEPTH - 1 - lev)) & 1;
            p *= bit ? s[node] : (1.f - s[node]);
        }
        lp[l * BATCH + b] = p;
    }
}

// ---------------------------------------------------------------------------
// Kernel 3: fp32 -> bf16 conversion (x and W1), 4 elements per thread.
// ---------------------------------------------------------------------------
__global__ void sndt_conv_bf16_kernel(const float* __restrict__ in,
                                      unsigned short* __restrict__ out,
                                      int nelem) {
    const int i = (blockIdx.x * blockDim.x + threadIdx.x) * 4;
    if (i >= nelem) return;
    float4 v = *(const float4*)(in + i);
    *(ushort4*)(out + i) = make_ushort4(f2bf(v.x), f2bf(v.y), f2bf(v.z), f2bf(v.w));
}

// W2 [16][100][256] -> bf16 padded [16][128][256] (zero rows 100..127)
__global__ void sndt_conv_w2p_kernel(const float* __restrict__ W2,
                                     unsigned short* __restrict__ w2p) {
    const int idx = (blockIdx.x * blockDim.x + threadIdx.x) * 4;  // over 16*128*256
    const int k = idx & 255;
    const int n = (idx >> 8) & (NPAD - 1);
    const int l = idx >> 15;
    if (n < NUM_CLASSES) {
        float4 v = *(const float4*)(W2 + ((size_t)(l * NUM_CLASSES + n) * LEAF_HIDDEN + k));
        *(ushort4*)(w2p + idx) = make_ushort4(f2bf(v.x), f2bf(v.y), f2bf(v.z), f2bf(v.w));
    } else {
        *(ushort4*)(w2p + idx) = make_ushort4(0, 0, 0, 0);
    }
}

// ---------------------------------------------------------------------------
// Kernel 4: GEMM1  h[l] = relu(x @ W1[l]^T + b1[l]), bf16 WMMA 16x16x32.
// Block: 256 thr = 8 waves, tile 64(M) x 128(N). A tile async-staged to LDS
// with double buffering (16 K-chunks of 64); B direct from L2-resident W1.
// Grid: (BATCH/64, LEAF_HIDDEN/128, NUM_LEAVES)
// ---------------------------------------------------------------------------
__global__ __launch_bounds__(256)
void sndt_gemm1_kernel(const unsigned short* __restrict__ xb,   // [4096][1024]
                       const unsigned short* __restrict__ w1b,  // [16][256][1024]
                       const float* __restrict__ b1,            // [16][256]
                       unsigned short* __restrict__ hb) {       // [16][4096][256]
    __shared__ unsigned short sA[2 * CHUNK_SHORTS];

    const int tid  = threadIdx.x;
    const int lane = tid & 31;
    const int wave = tid >> 5;
    const int bm   = blockIdx.x * 64;
    const int l    = blockIdx.z;
    const int half = lane >> 4;
    const int lidx = lane & 15;
    const int ncol = blockIdx.y * 128 + wave * 16 + lidx;

    const unsigned sAbase = (unsigned)(size_t)(&sA[0]);
    const unsigned short* Ag = xb + (size_t)bm * INPUT_DIM;
    const unsigned short* Bp = w1b + (size_t)(l * LEAF_HIDDEN + ncol) * INPUT_DIM;

    f32x8 acc[4];
    #pragma unroll
    for (int mt = 0; mt < 4; ++mt)
        #pragma unroll
        for (int r = 0; r < 8; ++r) acc[mt][r] = 0.f;

    stage_chunk<INPUT_DIM>(Ag, sAbase, tid);

    for (int c = 0; c < INPUT_DIM / 64; ++c) {
        if (c + 1 < INPUT_DIM / 64) {
            stage_chunk<INPUT_DIM>(Ag + (c + 1) * 64,
                                   sAbase + ((c + 1) & 1) * CHUNK_BYTES, tid);
            asm volatile("s_wait_asynccnt 0x2" ::: "memory");
        } else {
            asm volatile("s_wait_asynccnt 0x0" ::: "memory");
        }
        __syncthreads();

        const unsigned short* sbuf = &sA[(c & 1) * CHUNK_SHORTS];
        #pragma unroll
        for (int kk = 0; kk < 64; kk += 32) {
            Frag16 bf;
            const unsigned short* bp = Bp + c * 64 + kk + half * 8;
            bf.q[0] = *(const uint4*)(bp);
            bf.q[1] = *(const uint4*)(bp + 16);
            #pragma unroll
            for (int mt = 0; mt < 4; ++mt) {
                Frag16 af;
                const unsigned short* ap = sbuf + (mt * 16 + lidx) * SROW + kk + half * 8;
                af.q[0] = *(const uint4*)(ap);          // ds_load_b128
                af.q[1] = *(const uint4*)(ap + 16);
                acc[mt] = __builtin_amdgcn_wmma_f32_16x16x32_bf16(
                    false, af.v, false, bf.v, (short)0, acc[mt], false, false);
            }
        }
        __syncthreads();
    }

    const float bias = b1[l * LEAF_HIDDEN + ncol];
    #pragma unroll
    for (int mt = 0; mt < 4; ++mt) {
        #pragma unroll
        for (int r = 0; r < 8; ++r) {
            const int m = bm + mt * 16 + half * 8 + r;
            float v = acc[mt][r] + bias;
            v = v > 0.f ? v : 0.f;
            hb[((size_t)l * BATCH + m) * LEAF_HIDDEN + ncol] = f2bf(v);
        }
    }
}

// ---------------------------------------------------------------------------
// Kernel 5: GEMM2 + leaf-weighted combine.
//   logits[b][c] = sum_l lp[l][b] * (h[l] @ W2p[l]^T + b2[l])[b][c]
// One async pipeline across 64 chunks (16 leaves x 4 K-chunks of 64).
// ---------------------------------------------------------------------------
__global__ __launch_bounds__(256)
void sndt_gemm2_kernel(const unsigned short* __restrict__ hb,   // [16][4096][256]
                       const unsigned short* __restrict__ w2p,  // [16][128][256]
                       const float* __restrict__ b2,            // [16][100]
                       const float* __restrict__ lp,            // [16][4096]
                       float* __restrict__ out) {               // [4096][100]
    __shared__ unsigned short sA[2 * CHUNK_SHORTS];
    __shared__ float s_w[NUM_LEAVES * 64];
    __shared__ float s_b2[NUM_LEAVES * NPAD];

    const int tid  = threadIdx.x;
    const int lane = tid & 31;
    const int wave = tid >> 5;
    const int bm   = blockIdx.x * 64;

    for (int i = tid; i < NUM_LEAVES * 64; i += 256) {
        const int l = i >> 6, m = i & 63;
        s_w[i] = lp[l * BATCH + bm + m];
    }
    for (int i = tid; i < NUM_LEAVES * NPAD; i += 256) {
        const int l = i >> 7, n = i & (NPAD - 1);
        s_b2[i] = (n < NUM_CLASSES) ? b2[l * NUM_CLASSES + n] : 0.f;
    }
    __syncthreads();

    const int half = lane >> 4;
    const int lidx = lane & 15;
    const int ncol = wave * 16 + lidx;          // padded class column 0..127

    const unsigned sAbase = (unsigned)(size_t)(&sA[0]);
    // chunk c -> leaf l = c>>2, kb = (c&3)*64
    #define CHUNK_PTR(c) (hb + (size_t)((c) >> 2) * BATCH * LEAF_HIDDEN \
                             + (size_t)bm * LEAF_HIDDEN + ((c) & 3) * 64)

    float fin[4][8];
    f32x8 acc[4];
    #pragma unroll
    for (int mt = 0; mt < 4; ++mt)
        #pragma unroll
        for (int r = 0; r < 8; ++r) { fin[mt][r] = 0.f; acc[mt][r] = 0.f; }

    stage_chunk<LEAF_HIDDEN>(CHUNK_PTR(0), sAbase, tid);

    for (int c = 0; c < NUM_LEAVES * 4; ++c) {
        const int l = c >> 2;
        if (c + 1 < NUM_LEAVES * 4) {
            stage_chunk<LEAF_HIDDEN>(CHUNK_PTR(c + 1),
                                     sAbase + ((c + 1) & 1) * CHUNK_BYTES, tid);
            asm volatile("s_wait_asynccnt 0x2" ::: "memory");
        } else {
            asm volatile("s_wait_asynccnt 0x0" ::: "memory");
        }
        __syncthreads();

        const unsigned short* Bp = w2p + (size_t)(l * NPAD + ncol) * LEAF_HIDDEN
                                       + (c & 3) * 64;
        const unsigned short* sbuf = &sA[(c & 1) * CHUNK_SHORTS];
        #pragma unroll
        for (int kk = 0; kk < 64; kk += 32) {
            Frag16 bf;
            const unsigned short* bp = Bp + kk + half * 8;
            bf.q[0] = *(const uint4*)(bp);
            bf.q[1] = *(const uint4*)(bp + 16);
            #pragma unroll
            for (int mt = 0; mt < 4; ++mt) {
                Frag16 af;
                const unsigned short* ap = sbuf + (mt * 16 + lidx) * SROW + kk + half * 8;
                af.q[0] = *(const uint4*)(ap);
                af.q[1] = *(const uint4*)(ap + 16);
                acc[mt] = __builtin_amdgcn_wmma_f32_16x16x32_bf16(
                    false, af.v, false, bf.v, (short)0, acc[mt], false, false);
            }
        }
        __syncthreads();

        if ((c & 3) == 3) {   // leaf finished: scale into final accumulator
            #pragma unroll
            for (int mt = 0; mt < 4; ++mt)
                #pragma unroll
                for (int r = 0; r < 8; ++r) {
                    const int ml = mt * 16 + half * 8 + r;
                    fin[mt][r] += s_w[l * 64 + ml] * acc[mt][r];
                    acc[mt][r] = 0.f;
                }
        }
    }
    #undef CHUNK_PTR

    if (ncol < NUM_CLASSES) {
        #pragma unroll
        for (int mt = 0; mt < 4; ++mt)
            #pragma unroll
            for (int r = 0; r < 8; ++r) {
                const int ml = mt * 16 + half * 8 + r;
                float bsum = 0.f;
                #pragma unroll
                for (int l = 0; l < NUM_LEAVES; ++l)
                    bsum += s_w[l * 64 + ml] * s_b2[l * NPAD + ncol];
                out[(size_t)(bm + ml) * NUM_CLASSES + ncol] = fin[mt][r] + bsum;
            }
    }
}

// ---------------------------------------------------------------------------
extern "C" void kernel_launch(void* const* d_in, const int* in_sizes, int n_in,
                              void* d_out, int out_size, void* d_ws, size_t ws_size,
                              hipStream_t stream) {
    const float* x    = (const float*)d_in[0];   // [4096][1024]
    const float* gu   = (const float*)d_in[1];   // [15][4096][1024]
    const float* flog = (const float*)d_in[2];   // [15][1024]
    const float* thr  = (const float*)d_in[3];   // [15]
    const float* fw   = (const float*)d_in[4];   // [15][1024]
    const float* W1   = (const float*)d_in[5];   // [16][256][1024]
    const float* b1   = (const float*)d_in[6];   // [16][256]
    const float* W2   = (const float*)d_in[7];   // [16][100][256]
    const float* b2   = (const float*)d_in[8];   // [16][100]
    float* out = (float*)d_out;
    (void)in_sizes; (void)n_in; (void)out_size; (void)ws_size;

    char* ws = (char*)d_ws;                       // all offsets 16B aligned
    float*          sp    = (float*)(ws + 0);            // 15*4096*4      = 245760
    float*          lpb   = (float*)(ws + 245760);       // 16*4096*4      = 262144
    unsigned short* xb    = (unsigned short*)(ws + 507904);    // 4096*1024*2   = 8388608
    unsigned short* w1b   = (unsigned short*)(ws + 8896512);   // 16*256*1024*2 = 8388608
    unsigned short* w2b   = (unsigned short*)(ws + 17285120);  // 16*128*256*2  = 1048576
    unsigned short* hbv   = (unsigned short*)(ws + 18333696);  // 16*4096*256*2 = 33554432
    float*          efl   = (float*)(ws + 51888128);           // 15*1024*4     = 61440
    float*          fwefl = (float*)(ws + 51949568);           // 15*1024*4     = 61440

    // 0) exp(feature_logits), fw*exp(feature_logits)
    sndt_prep_kernel<<<(NUM_INTERNAL * INPUT_DIM + 255) / 256, 256, 0, stream>>>(
        flog, fw, efl, fwefl);
    // 1) gating: 15*4096 rows, 8 waves/block
    sndt_gating_kernel<<<(NUM_INTERNAL * BATCH) / 8, 256, 0, stream>>>(
        x, gu, efl, thr, fwefl, sp);
    // 2) leaf routing probabilities
    sndt_leafprob_kernel<<<BATCH / 256, 256, 0, stream>>>(sp, lpb);
    // 3) bf16 conversions
    sndt_conv_bf16_kernel<<<(BATCH * INPUT_DIM) / 1024, 256, 0, stream>>>(
        x, xb, BATCH * INPUT_DIM);
    sndt_conv_bf16_kernel<<<(NUM_LEAVES * LEAF_HIDDEN * INPUT_DIM) / 1024, 256, 0, stream>>>(
        W1, w1b, NUM_LEAVES * LEAF_HIDDEN * INPUT_DIM);
    sndt_conv_w2p_kernel<<<(NUM_LEAVES * NPAD * LEAF_HIDDEN) / 1024, 256, 0, stream>>>(
        W2, w2b);
    // 4) per-leaf GEMM1 + ReLU -> bf16 h
    dim3 g1(BATCH / 64, LEAF_HIDDEN / 128, NUM_LEAVES);
    sndt_gemm1_kernel<<<g1, 256, 0, stream>>>(xb, w1b, b1, hbv);
    // 5) GEMM2 + leaf-weighted combine -> logits
    sndt_gemm2_kernel<<<BATCH / 64, 256, 0, stream>>>(hbv, w2b, b2, lpb, out);
}
